// PyramidAttention_89026082111944
// MI455X (gfx1250) — matrix operand order repaired and checked by reading, not compiled, not run on hardware
//
#include <hip/hip_runtime.h>
#include <hip/hip_bf16.h>
#include <stdint.h>

// ---------------- constants ----------------
#define NB    4
#define CCH   64      // channels
#define CR    32      // reduced channels
#define H0    48
#define HW0   2304    // 48*48
#define LTOT  7470    // 48^2+43^2+38^2+33^2+28^2
#define LPAD  7488    // 234 * 32
#define LFULL 7456    // 233 full 32-key chunks; epilogue covers [7456, 7470)
#define LDBL  7424    // 116 double (64-key) iterations
#define NEG_BIG (-1e30f)

typedef __attribute__((ext_vector_type(16))) __bf16 v16bf;
typedef __attribute__((ext_vector_type(8)))  float  v8f;

union FragAB { v16bf v; uint4 q[2]; };

__device__ __forceinline__ unsigned short f2bf(float f) {
  union { float f; unsigned u; } x; x.f = f;
  unsigned r = x.u + 0x7FFFu + ((x.u >> 16) & 1u);   // RNE to bf16
  return (unsigned short)(r >> 16);
}
// pack two f32 -> two bf16 (round-nearest via +0x8000, then byte-select): 3 VALU ops
__device__ __forceinline__ unsigned pack2bf(float lo, float hi) {
  union { float f; unsigned u; } a, b; a.f = lo; b.f = hi;
  return __builtin_amdgcn_perm(b.u + 0x8000u, a.u + 0x8000u, 0x07060302u);
}

// ---------------- bicubic (a = -0.75, align_corners=False, replicate) ----------------
__device__ __forceinline__ float cubicw(float d) {
  const float a = -0.75f;
  d = fabsf(d);
  if (d <= 1.0f)      return ((a + 2.0f) * d - (a + 3.0f)) * d * d + 1.0f;
  else if (d < 2.0f)  return (((a * d) - 5.0f * a) * d + 8.0f * a) * d - 4.0f * a;
  return 0.0f;
}

__global__ void bicubic_kernel(const float* __restrict__ x, float* __restrict__ dst,
                               int oh, int ow) {
  int idx = blockIdx.x * blockDim.x + threadIdx.x;
  int total = NB * CCH * oh * ow;
  if (idx >= total) return;
  int wo = idx % ow; int t = idx / ow;
  int ho = t % oh;   int nc = t / oh;
  float sh = 48.0f / (float)oh, sw = 48.0f / (float)ow;
  float fy = ((float)ho + 0.5f) * sh - 0.5f;
  float fx = ((float)wo + 0.5f) * sw - 0.5f;
  int y0 = (int)floorf(fy); float ty = fy - (float)y0;
  int x0 = (int)floorf(fx); float tx = fx - (float)x0;
  float wy[4], wx[4];
#pragma unroll
  for (int o = 0; o < 4; ++o) { wy[o] = cubicw(ty - (float)(o - 1)); wx[o] = cubicw(tx - (float)(o - 1)); }
  const float* src = x + (long)nc * (H0 * H0);
  float acc = 0.0f;
#pragma unroll
  for (int oy = 0; oy < 4; ++oy) {
    int yy = min(max(y0 + oy - 1, 0), H0 - 1);
#pragma unroll
    for (int ox = 0; ox < 4; ++ox) {
      int xx = min(max(x0 + ox - 1, 0), H0 - 1);
      acc += wy[oy] * wx[ox] * src[yy * H0 + xx];
    }
  }
  dst[idx] = acc;
}

// ---------------- conv3x3 (pad 1), writes bf16 row-major [rowBase+l][CR] ----------------
__global__ void conv3x3_bf16_kernel(const float* __restrict__ src, int Hs, int Ws,
                                    const float* __restrict__ wgt, const float* __restrict__ bias,
                                    unsigned short* __restrict__ dst, long nStrideOut, long rowBase) {
  int l = blockIdx.x * blockDim.x + threadIdx.x;
  int HW = Hs * Ws;
  if (l >= HW) return;
  int co = blockIdx.y, n = blockIdx.z;
  int h = l / Ws, w = l % Ws;
  const float* sp = src + (long)n * CCH * HW;
  const float* wp = wgt + (long)co * CCH * 9;
  float s = bias[co];
  for (int ci = 0; ci < CCH; ++ci) {
    const float* spc = sp + (long)ci * HW;
    const float* wpc = wp + ci * 9;
#pragma unroll
    for (int kh = 0; kh < 3; ++kh) {
      int hh = h + kh - 1;
      if ((unsigned)hh >= (unsigned)Hs) continue;
#pragma unroll
      for (int kw = 0; kw < 3; ++kw) {
        int ww = w + kw - 1;
        if ((unsigned)ww >= (unsigned)Ws) continue;
        s += wpc[kh * 3 + kw] * spc[hh * Ws + ww];
      }
    }
  }
  dst[(long)n * nStrideOut + (rowBase + l) * CR + co] = f2bf(s);
}

// ---------------- conv1x1, writes bf16 channel-major [n][64][LPAD] at colBase ----------------
__global__ void conv1x1_bf16_kernel(const float* __restrict__ src, int HW,
                                    const float* __restrict__ wgt, const float* __restrict__ bias,
                                    unsigned short* __restrict__ dst, long colBase) {
  int l = blockIdx.x * blockDim.x + threadIdx.x;
  if (l >= HW) return;
  int co = blockIdx.y, n = blockIdx.z;
  const float* sp = src + (long)n * CCH * HW;
  const float* wp = wgt + co * CCH;
  float s = bias[co];
  for (int ci = 0; ci < CCH; ++ci) s += wp[ci] * sp[(long)ci * HW + l];
  dst[((long)n * CCH + co) * LPAD + colBase + l] = f2bf(s);
}

// ---------------- zero the L padding of K and V ----------------
__global__ void pad_zero_kernel(unsigned short* __restrict__ kt, unsigned short* __restrict__ v) {
  int tid = blockIdx.x * blockDim.x + threadIdx.x;
  const int PADL = LPAD - LTOT;                 // 18
  int nk = NB * PADL * CR;
  if (tid < nk) {
    int c = tid % CR; int r = tid / CR; int row = r % PADL; int n = r / PADL;
    kt[((long)n * LPAD + LTOT + row) * CR + c] = 0;
  }
  int nv = NB * CCH * PADL;
  if (tid < nv) {
    int col = tid % PADL; int t = tid / PADL; int c = t % CCH; int n = t / CCH;
    v[((long)n * CCH + c) * LPAD + LTOT + col] = 0;
  }
}

// ---------------- fused flash attention (transposed score tiles) ----------------
__device__ __forceinline__ float max8(const v8f& s) {
  float a0 = fmaxf(s[0], s[1]), a1 = fmaxf(s[2], s[3]);
  float a2 = fmaxf(s[4], s[5]), a3 = fmaxf(s[6], s[7]);
  return fmaxf(fmaxf(a0, a1), fmaxf(a2, a3));
}

// One 32-key chunk: S^T = K*Q^T (2 WMMA), online softmax, P^T restage, Y^T += V*P^T (4 WMMA).
__device__ __forceinline__ void chunk_step(
    const unsigned short* __restrict__ Kn, const unsigned short* __restrict__ Vn,
    const FragAB& bQ, int l0, unsigned short* sbase /* parity-selected [16][40] */,
    int col, int half, float& m_st, float& l_st, v8f (&acc)[4], const v8f& z8) {
  // A-fragments of K (lane = key row within subtile)
  FragAB aK0, aK1;
  {
    const uint4* kp0 = (const uint4*)(Kn + (long)(l0 + col) * CR);
    aK0.q[0] = kp0[half]; aK0.q[1] = kp0[2 + half];
    const uint4* kp1 = (const uint4*)(Kn + (long)(l0 + 16 + col) * CR);
    aK1.q[0] = kp1[half]; aK1.q[1] = kp1[2 + half];
  }
  // S^T tiles: element r <-> key l0 + st*16 + half*8 + r, query q0+col
  v8f S0 = __builtin_amdgcn_wmma_f32_16x16x32_bf16(false, aK0.v, false, bQ.v, (short)0, z8, false, false);
  v8f S1 = __builtin_amdgcn_wmma_f32_16x16x32_bf16(false, aK1.v, false, bQ.v, (short)0, z8, false, false);

  // ---- online softmax: reduction dim is in-lane + one half-swap ----
  float cm = fmaxf(max8(S0), max8(S1));
  cm = fmaxf(cm, __shfl_xor(cm, 16, 32));
  float mnew  = fmaxf(m_st, cm);
  float alpha = __expf(m_st - mnew);
  float p0[8], p1[8];
#pragma unroll
  for (int r = 0; r < 8; ++r) { p0[r] = __expf(S0[r] - mnew); p1[r] = __expf(S1[r] - mnew); }
  float rs = ((p0[0] + p0[1]) + (p0[2] + p0[3])) + ((p0[4] + p0[5]) + (p0[6] + p0[7]))
           + ((p1[0] + p1[1]) + (p1[2] + p1[3])) + ((p1[4] + p1[5]) + (p1[6] + p1[7]));
  rs += __shfl_xor(rs, 16, 32);
  l_st = l_st * alpha + rs;
  m_st = mnew;
#pragma unroll
  for (int t = 0; t < 4; ++t)
#pragma unroll
    for (int i = 0; i < 8; ++i) acc[t][i] *= alpha;

  // ---- restage P^T as [query][key] (packed b32 stores, v_perm pack) ----
  unsigned* spw = (unsigned*)(sbase + col * 40);
#pragma unroll
  for (int k = 0; k < 4; ++k) {
    spw[half * 4 + k]     = pack2bf(p0[2 * k], p0[2 * k + 1]);   // keys  half*8+2k
    spw[8 + half * 4 + k] = pack2bf(p1[2 * k], p1[2 * k + 1]);   // keys 16+half*8+2k
  }
  FragAB bP;
  {
    const uint4* spr = (const uint4*)(sbase + col * 40);
    bP.q[0] = spr[half]; bP.q[1] = spr[2 + half];
  }

  // ---- Y^T += V * P^T over 4 channel tiles ----
#pragma unroll
  for (int tt = 0; tt < 4; ++tt) {
    FragAB aV;
    const uint4* vp = (const uint4*)(Vn + (long)(tt * 16 + col) * LPAD + l0);
    aV.q[0] = vp[half]; aV.q[1] = vp[2 + half];
    acc[tt] = __builtin_amdgcn_wmma_f32_16x16x32_bf16(false, aV.v, false, bP.v, (short)0, acc[tt], false, false);
  }
}

__global__ void __launch_bounds__(128)
flash_attn_kernel(const unsigned short* __restrict__ Q,
                  const unsigned short* __restrict__ K,
                  const unsigned short* __restrict__ V,
                  const float* __restrict__ xres,
                  float* __restrict__ out) {
  // P^T staging, double-buffered per wave: [wave][parity][query 16][key 32 + pad]
  __shared__ __align__(16) unsigned short sP[4][2][16][40];
  const int lane = threadIdx.x;
  const int wy   = threadIdx.y;
  const int n    = blockIdx.y;
  const int q0   = (blockIdx.x * 4 + wy) * 16;
  const int col  = lane & 15;
  const int half = lane >> 4;

  // ---- B-fragment of Q (constant over chunks): lane = query column ----
  FragAB bQ;
  {
    const uint4* qp = (const uint4*)(Q + ((long)n * HW0 + q0 + col) * CR);
    bQ.q[0] = qp[half]; bQ.q[1] = qp[2 + half];
  }

  float m_st = NEG_BIG, l_st = 0.0f;
  v8f acc[4], z8;
#pragma unroll
  for (int i = 0; i < 8; ++i) z8[i] = 0.0f;
#pragma unroll
  for (int t = 0; t < 4; ++t)
#pragma unroll
    for (int i = 0; i < 8; ++i) acc[t][i] = 0.0f;

  const unsigned short* Kn = K + (long)n * LPAD * CR;
  const unsigned short* Vn = V + (long)n * CCH * LPAD;
  unsigned short* sb0 = &sP[wy][0][0][0];
  unsigned short* sb1 = &sP[wy][1][0][0];

  // ===== main loop: two independent 32-key chunks per iteration (fills WMMA hazard slots) =====
  for (int l0 = 0; l0 < LDBL; l0 += 64) {
    chunk_step(Kn, Vn, bQ, l0,      sb0, col, half, m_st, l_st, acc, z8);
    chunk_step(Kn, Vn, bQ, l0 + 32, sb1, col, half, m_st, l_st, acc, z8);
  }
  // remaining full chunk [7424, 7456)
  chunk_step(Kn, Vn, bQ, LDBL, sb0, col, half, m_st, l_st, acc, z8);

  // ===== epilogue: keys [7456, 7470), one masked subtile =====
  {
    const int l0 = LFULL;
    FragAB aK0;
    {
      const uint4* kp0 = (const uint4*)(Kn + (long)(l0 + col) * CR);
      aK0.q[0] = kp0[half]; aK0.q[1] = kp0[2 + half];
    }
    v8f S0 = __builtin_amdgcn_wmma_f32_16x16x32_bf16(false, aK0.v, false, bQ.v, (short)0, z8, false, false);
#pragma unroll
    for (int r = 0; r < 8; ++r)
      if (half * 8 + r >= (LTOT - LFULL)) S0[r] = NEG_BIG;   // mask keys >= 7470

    float cm = max8(S0);
    cm = fmaxf(cm, __shfl_xor(cm, 16, 32));
    float mnew  = fmaxf(m_st, cm);
    float alpha = __expf(m_st - mnew);
    float p0[8];
#pragma unroll
    for (int r = 0; r < 8; ++r) p0[r] = __expf(S0[r] - mnew);
    float rs = ((p0[0] + p0[1]) + (p0[2] + p0[3])) + ((p0[4] + p0[5]) + (p0[6] + p0[7]));
    rs += __shfl_xor(rs, 16, 32);
    l_st = l_st * alpha + rs;
#pragma unroll
    for (int t = 0; t < 4; ++t)
#pragma unroll
      for (int i = 0; i < 8; ++i) acc[t][i] *= alpha;

    unsigned* spw = (unsigned*)(sb1 + col * 40);
#pragma unroll
    for (int k = 0; k < 4; ++k) {
      spw[half * 4 + k]     = pack2bf(p0[2 * k], p0[2 * k + 1]);
      spw[8 + half * 4 + k] = 0u;                               // keys 16..31: P = 0 (V pad is zeroed)
    }
    FragAB bP;
    {
      const uint4* spr = (const uint4*)(sb1 + col * 40);
      bP.q[0] = spr[half]; bP.q[1] = spr[2 + half];
    }
#pragma unroll
    for (int tt = 0; tt < 4; ++tt) {
      FragAB aV;
      const uint4* vp = (const uint4*)(Vn + (long)(tt * 16 + col) * LPAD + l0);
      aV.q[0] = vp[half]; aV.q[1] = vp[2 + half];
      acc[tt] = __builtin_amdgcn_wmma_f32_16x16x32_bf16(false, aV.v, false, bP.v, (short)0, acc[tt], false, false);
    }
  }

  // ---- finalize: Y^T tile is [channel rows x query cols]; q = q0+col per lane ----
  float linv = 1.0f / l_st;
  const int q = q0 + col;
#pragma unroll
  for (int tt = 0; tt < 4; ++tt) {
#pragma unroll
    for (int r = 0; r < 8; ++r) {
      int c = tt * 16 + half * 8 + r;
      long o = ((long)n * CCH + c) * HW0 + q;
      out[o] = acc[tt][r] * linv + xres[o];
    }
  }
}

// ---------------- host ----------------
extern "C" void kernel_launch(void* const* d_in, const int* in_sizes, int n_in,
                              void* d_out, int out_size, void* d_ws, size_t ws_size,
                              hipStream_t stream) {
  (void)in_sizes; (void)n_in; (void)out_size; (void)ws_size;
  const float* x  = (const float*)d_in[0];
  const float* wb = (const float*)d_in[1];
  const float* bb = (const float*)d_in[2];
  const float* wm = (const float*)d_in[3];
  const float* bm = (const float*)d_in[4];
  const float* wa = (const float*)d_in[5];
  const float* ba = (const float*)d_in[6];
  float* out = (float*)d_out;

  static const int  SH[5]    = {48, 43, 38, 33, 28};
  static const long LBASE[5] = {0, 2304, 4153, 5597, 6686};

  char* ws = (char*)d_ws;
  size_t off = 0;
  float* refbuf[4];
  for (int s = 1; s < 5; ++s) {
    refbuf[s - 1] = (float*)(ws + off);
    off += (size_t)NB * CCH * SH[s] * SH[s] * sizeof(float);
  }
  unsigned short* Qb = (unsigned short*)(ws + off); off += (size_t)NB * HW0 * CR * 2;
  unsigned short* Kb = (unsigned short*)(ws + off); off += (size_t)NB * LPAD * CR * 2;
  unsigned short* Vb = (unsigned short*)(ws + off); off += (size_t)NB * CCH * LPAD * 2;

  // 1) bicubic pyramid levels (scales 0.9 .. 0.6)
  for (int s = 1; s < 5; ++s) {
    int total = NB * CCH * SH[s] * SH[s];
    bicubic_kernel<<<(total + 255) / 256, 256, 0, stream>>>(x, refbuf[s - 1], SH[s], SH[s]);
  }

  // 2) queries: conv3x3(x, wb) -> bf16 [N, HW0, 32]
  {
    dim3 g((HW0 + 127) / 128, CR, NB);
    conv3x3_bf16_kernel<<<g, 128, 0, stream>>>(x, H0, H0, wb, bb, Qb, (long)HW0 * CR, 0L);
  }
  // 3) keys per level -> bf16 key-major [N, LPAD, 32]; values -> bf16 channel-major [N, 64, LPAD]
  for (int s = 0; s < 5; ++s) {
    const float* src = (s == 0) ? x : refbuf[s - 1];
    int Hs = SH[s], HW = Hs * Hs;
    dim3 gk((HW + 127) / 128, CR, NB);
    conv3x3_bf16_kernel<<<gk, 128, 0, stream>>>(src, Hs, Hs, wm, bm, Kb, (long)LPAD * CR, LBASE[s]);
    dim3 gv((HW + 127) / 128, CCH, NB);
    conv1x1_bf16_kernel<<<gv, 128, 0, stream>>>(src, HW, wa, ba, Vb, LBASE[s]);
  }

  // 4) zero the L padding (V padding required so epilogue WMMA sees 0*0, never NaN)
  {
    int nmax = NB * CCH * (LPAD - LTOT);
    pad_zero_kernel<<<(nmax + 255) / 256, 256, 0, stream>>>(Kb, Vb);
  }

  // 5) fused flash attention + residual
  {
    dim3 g(HW0 / 64, NB);   // 36 x 4 workgroups, 4 waves each, 1 wave = 16 queries
    dim3 b(32, 4);
    flash_attn_kernel<<<g, b, 0, stream>>>(Qb, Kb, Vb, x, out);
  }
}
